// GATBlock_41901700940059
// MI455X (gfx1250) — compile-verified, matrix operand order checked by reading, not run on hardware
//
#include <hip/hip_runtime.h>
#include <hip/hip_bf16.h>
#include <stdint.h>

// Problem constants (match reference)
#define NN   50000
#define EE   400000
#define FF   1040
#define HH   8
#define CC   130
#define ETOT (EE + NN)
#define KP   1056          // K padded to multiple of 32 (33 * 32)
#define MP   50048         // rows padded to multiple of 64 (64 * 782)
#define FP   1088          // out-features padded to multiple of 64 (64 * 17); xh row stride
#define MSUP 782           // MP / 64
#define NGRP 17            // FP / 64
#define GEMM_TASKS (MSUP * NGRP)

typedef __attribute__((ext_vector_type(16))) __bf16 v16bf;
typedef __attribute__((ext_vector_type(8)))  float  v8f;

// ---------- helpers ----------
__device__ __forceinline__ unsigned short f2bf(float f) {
    unsigned u = __float_as_uint(f);
    unsigned r = u + 0x7FFFu + ((u >> 16) & 1u);   // round-to-nearest-even
    return (unsigned short)(r >> 16);
}

__device__ __forceinline__ void atomicMaxF(float* addr, float val) {
    unsigned* ua = (unsigned*)addr;
    unsigned old = *ua;
    while (__uint_as_float(old) < val) {
        unsigned assumed = old;
        old = atomicCAS(ua, assumed, __float_as_uint(val));
        if (old == assumed) break;
    }
}

// ---------- 0: init scratch ----------
__global__ void k_init(float* __restrict__ agg, float* __restrict__ emax,
                       float* __restrict__ denom, float* __restrict__ stats) {
    size_t i = (size_t)blockIdx.x * blockDim.x + threadIdx.x;
    if (i < (size_t)NN * FF) agg[i] = 0.0f;
    if (i < (size_t)NN * HH) { emax[i] = -3.0e38f; denom[i] = 0.0f; }
    if (i < 2 * FF) stats[i] = 0.0f;
}

// ---------- 1: convert x -> bf16, M- and K-padded ----------
__global__ void k_conv_x(const float* __restrict__ x, unsigned short* __restrict__ xb) {
    size_t i = (size_t)blockIdx.x * blockDim.x + threadIdx.x;
    if (i >= (size_t)MP * KP) return;
    int k = (int)(i % KP);
    size_t n = i / KP;
    float v = (n < NN && k < FF) ? x[n * FF + k] : 0.0f;
    xb[i] = f2bf(v);
}

// ---------- 2: convert + transpose W -> Wt[n][k] bf16, N- and K-padded ----------
__global__ void k_conv_w(const float* __restrict__ W, unsigned short* __restrict__ wt) {
    size_t i = (size_t)blockIdx.x * blockDim.x + threadIdx.x;
    if (i >= (size_t)FP * KP) return;
    int k = (int)(i % KP);
    size_t n = i / KP;
    float v = (n < FF && k < FF) ? W[(size_t)k * FF + n] : 0.0f;
    wt[i] = f2bf(v);
}

// ---------- 3: GEMM xh = x @ W via v_wmma_f32_16x16x32_bf16 ----------
// One wave computes a 64x64 output block: 4 A fragments + 4 streamed B
// fragments -> 16 WMMAs per K-step. 128 acc + 32 A + 8 B VGPRs stays under
// 256 (no VGPR-MSB bank movs, no hazard nops, decent occupancy).
// g-major task mapping: the 8 waves of a block share the same 64 Wt rows.
__global__ void __launch_bounds__(256) k_gemm(const unsigned short* __restrict__ xb,
                                              const unsigned short* __restrict__ wt,
                                              float* __restrict__ xh) {
    const int wave = (int)((blockIdx.x * blockDim.x + threadIdx.x) >> 5);
    const int lane = threadIdx.x & 31;
    if (wave >= GEMM_TASKS) return;          // wave-uniform exit; EXEC stays all-1 for WMMA

    const int g      = wave / MSUP;          // column group: 64 cols
    const int mSuper = wave % MSUP;          // row block: 64 rows
    const int m0     = mSuper * 64;
    const int n0     = g * 64;
    const int row16  = lane & 15;
    const int koff   = (lane >> 4) * 8;      // ISA 16-bit A layout K-half interleave

    v8f acc[16];                             // acc[i*4+j] = tile (m0+i*16, n0+j*16)
#pragma unroll
    for (int t = 0; t < 16; ++t) acc[t] = (v8f){0.f,0.f,0.f,0.f,0.f,0.f,0.f,0.f};

    const unsigned short* aBase = xb + (size_t)(m0 + row16) * KP;
    const unsigned short* bBase = wt + (size_t)(n0 + row16) * KP;

    for (int k0 = 0; k0 < KP; k0 += 32) {
        __builtin_prefetch(aBase + k0 + 64, 0, 1);   // global_prefetch next K blocks
        union { v16bf v; uint4 u[2]; } A[4];
#pragma unroll
        for (int i = 0; i < 4; ++i) {
            const unsigned short* p = aBase + (size_t)i * 16 * KP + k0 + koff;
            A[i].u[0] = *(const uint4*)(p);
            A[i].u[1] = *(const uint4*)(p + 16);
        }
#pragma unroll
        for (int j = 0; j < 4; ++j) {
            const unsigned short* p = bBase + (size_t)j * 16 * KP + k0 + koff;
            union { v16bf v; uint4 u[2]; } B;
            B.u[0] = *(const uint4*)(p);
            B.u[1] = *(const uint4*)(p + 16);
#pragma unroll
            for (int i = 0; i < 4; ++i)
                acc[i * 4 + j] = __builtin_amdgcn_wmma_f32_16x16x32_bf16(
                    false, A[i].v, false, B.v, (short)0, acc[i * 4 + j], false, false);
        }
    }

    // C/D layout: VGPR r -> M = r + (lane>=16 ? 8 : 0), N = lane%16
    const int rbase = (lane >> 4) * 8;
    const int col   = lane & 15;
#pragma unroll
    for (int i = 0; i < 4; ++i) {
#pragma unroll
        for (int j = 0; j < 4; ++j) {
            float* orow = xh + (size_t)(m0 + i * 16) * FP + n0 + j * 16 + col;
#pragma unroll
            for (int r = 0; r < 8; ++r)
                orow[(size_t)(r + rbase) * FP] = acc[i * 4 + j][r];
        }
    }
}

// ---------- 4: per-node attention logits ----------
__global__ void k_attn(const float* __restrict__ xh, const float* __restrict__ att_src,
                       const float* __restrict__ att_dst,
                       float* __restrict__ asrc, float* __restrict__ adst) {
    int i = blockIdx.x * blockDim.x + threadIdx.x;   // over N*H
    if (i >= NN * HH) return;
    int h = i & 7;
    int n = i >> 3;
    const float* row = xh + (size_t)n * FP + h * CC;
    const float* as  = att_src + h * CC;
    const float* ad  = att_dst + h * CC;
    float s1 = 0.f, s2 = 0.f;
    for (int c = 0; c < CC; ++c) { float v = row[c]; s1 += v * as[c]; s2 += v * ad[c]; }
    asrc[i] = s1;
    adst[i] = s2;
}

// ---------- 5: edge pass 1 — leaky-relu logit + segment max ----------
__global__ void k_edge1(const int* __restrict__ ei, const float* __restrict__ asrc,
                        const float* __restrict__ adst,
                        float* __restrict__ ebuf, float* __restrict__ emax) {
    int i = blockIdx.x * blockDim.x + threadIdx.x;   // over ETOT*H
    if (i >= ETOT * HH) return;
    int h = i & 7;
    int t = i >> 3;
    int s = (t < EE) ? ei[t]      : (t - EE);        // self loops appended
    int d = (t < EE) ? ei[EE + t] : (t - EE);
    float e = asrc[s * HH + h] + adst[d * HH + h];
    e = (e > 0.f) ? e : 0.2f * e;                    // leaky_relu(0.2)
    ebuf[i] = e;
    atomicMaxF(&emax[d * HH + h], e);
}

// ---------- 6: edge pass 2 — exp and segment sum ----------
__global__ void k_edge2(const int* __restrict__ ei, float* __restrict__ ebuf,
                        const float* __restrict__ emax, float* __restrict__ denom) {
    int i = blockIdx.x * blockDim.x + threadIdx.x;
    if (i >= ETOT * HH) return;
    int h = i & 7;
    int t = i >> 3;
    int d = (t < EE) ? ei[EE + t] : (t - EE);
    float e2 = __expf(ebuf[i] - emax[d * HH + h]);
    ebuf[i] = e2;
    atomicAdd(&denom[d * HH + h], e2);
}

// ---------- 7: message scatter (wave per edge, global f32 atomics) ----------
__global__ void __launch_bounds__(256) k_scatter(const int* __restrict__ ei,
                                                 const float* __restrict__ xh,
                                                 const float* __restrict__ ebuf,
                                                 const float* __restrict__ denom,
                                                 float* __restrict__ agg) {
    int t = (int)((blockIdx.x * blockDim.x + threadIdx.x) >> 5);
    int lane = threadIdx.x & 31;
    if (t >= ETOT) return;
    int s = (t < EE) ? ei[t]      : (t - EE);
    int d = (t < EE) ? ei[EE + t] : (t - EE);
    const float* xrow = xh  + (size_t)s * FP;
    float*       orow = agg + (size_t)d * FF;
    float alpha[HH];
#pragma unroll
    for (int h = 0; h < HH; ++h)
        alpha[h] = ebuf[t * HH + h] / (denom[d * HH + h] + 1e-16f);
    for (int f = lane; f < FF; f += 32) {
        int h = f / CC;
        atomicAdd(&orow[f], alpha[h] * xrow[f]);     // global_atomic_add_f32
    }
}

// ---------- 8: y = prev + agg + bias ; two-level BN stats via ds_add_f32 ----------
#define ROWS_PER_BLK 50
__global__ void __launch_bounds__(256) k_ystats(const float* __restrict__ prev,
                                                const float* __restrict__ agg,
                                                const float* __restrict__ bias,
                                                float* __restrict__ y,
                                                float* __restrict__ stats) {
    __shared__ float s1[FF];
    __shared__ float s2[FF];
    for (int f = threadIdx.x; f < FF; f += 256) { s1[f] = 0.f; s2[f] = 0.f; }
    __syncthreads();
    int r0 = blockIdx.x * ROWS_PER_BLK;
    for (int idx = threadIdx.x; idx < ROWS_PER_BLK * FF; idx += 256) {
        int r = r0 + idx / FF;
        int f = idx % FF;
        size_t off = (size_t)r * FF + f;
        float v = prev[off] + agg[off] + bias[f];
        y[off] = v;
        atomicAdd(&s1[f], v);        // ds_add_f32
        atomicAdd(&s2[f], v * v);
    }
    __syncthreads();
    for (int f = threadIdx.x; f < FF; f += 256) {
        atomicAdd(&stats[f],      s1[f]);
        atomicAdd(&stats[FF + f], s2[f]);
    }
}

// ---------- 9: BN normalize + ReLU in place ----------
__global__ void k_bnfinal(float* __restrict__ y, const float* __restrict__ stats,
                          const float* __restrict__ gamma, const float* __restrict__ beta) {
    size_t i = (size_t)blockIdx.x * blockDim.x + threadIdx.x;
    if (i >= (size_t)NN * FF) return;
    int f = (int)(i % FF);
    const float invN = 1.0f / (float)NN;
    float mean = stats[f] * invN;
    float var  = stats[FF + f] * invN - mean * mean;
    float v = (y[i] - mean) * rsqrtf(var + 1e-5f) * gamma[f] + beta[f];
    y[i] = v > 0.f ? v : 0.f;
}

// ---------- host ----------
static inline size_t alignup(size_t x) { return (x + 255) & ~(size_t)255; }

extern "C" void kernel_launch(void* const* d_in, const int* in_sizes, int n_in,
                              void* d_out, int out_size, void* d_ws, size_t ws_size,
                              hipStream_t stream) {
    const float* prev    = (const float*)d_in[0];
    const float* x       = (const float*)d_in[1];
    const int*   ei      = (const int*)  d_in[2];
    const float* W       = (const float*)d_in[3];
    const float* att_src = (const float*)d_in[4];
    const float* att_dst = (const float*)d_in[5];
    const float* bias    = (const float*)d_in[6];
    const float* gamma   = (const float*)d_in[7];
    const float* beta    = (const float*)d_in[8];
    float* y = (float*)d_out;

    // workspace carve-out
    char* ws = (char*)d_ws;
    size_t off = 0;
    unsigned short* xb = (unsigned short*)(ws + off); off += alignup((size_t)MP * KP * 2);
    unsigned short* wt = (unsigned short*)(ws + off); off += alignup((size_t)FP * KP * 2);
    float* xh    = (float*)(ws + off); off += alignup((size_t)MP * FP * 4);
    float* agg   = (float*)(ws + off); off += alignup((size_t)NN * FF * 4);
    float* asrc  = (float*)(ws + off); off += alignup((size_t)NN * HH * 4);
    float* adst  = (float*)(ws + off); off += alignup((size_t)NN * HH * 4);
    float* emax  = (float*)(ws + off); off += alignup((size_t)NN * HH * 4);
    float* denom = (float*)(ws + off); off += alignup((size_t)NN * HH * 4);
    float* ebuf  = (float*)(ws + off); off += alignup((size_t)ETOT * HH * 4);
    float* stats = (float*)(ws + off); off += alignup((size_t)2 * FF * 4);
    (void)ws_size; (void)n_in; (void)in_sizes; (void)out_size;

    const int TPB = 256;
    dim3 blk(TPB);

    // 0: zero/init scratch (covers largest extent)
    {
        size_t n = (size_t)NN * FF;
        k_init<<<dim3((unsigned)((n + TPB - 1) / TPB)), blk, 0, stream>>>(agg, emax, denom, stats);
    }
    // 1: x -> bf16, padded [MP][KP]
    {
        size_t n = (size_t)MP * KP;
        k_conv_x<<<dim3((unsigned)((n + TPB - 1) / TPB)), blk, 0, stream>>>(x, xb);
    }
    // 2: W -> bf16 transposed, padded [FP][KP]
    {
        size_t n = (size_t)FP * KP;
        k_conv_w<<<dim3((unsigned)((n + TPB - 1) / TPB)), blk, 0, stream>>>(W, wt);
    }
    // 3: WMMA GEMM (64x64 per wave, 8 waves per block)
    {
        unsigned nblk = (GEMM_TASKS + 7) / 8;
        k_gemm<<<dim3(nblk), blk, 0, stream>>>(xb, wt, xh);
    }
    // 4: attention logits per node
    k_attn<<<dim3((NN * HH + TPB - 1) / TPB), blk, 0, stream>>>(xh, att_src, att_dst, asrc, adst);
    // 5/6: edge softmax
    k_edge1<<<dim3((ETOT * HH + TPB - 1) / TPB), blk, 0, stream>>>(ei, asrc, adst, ebuf, emax);
    k_edge2<<<dim3((ETOT * HH + TPB - 1) / TPB), blk, 0, stream>>>(ei, ebuf, emax, denom);
    // 7: weighted message scatter
    k_scatter<<<dim3((ETOT + 7) / 8), blk, 0, stream>>>(ei, xh, ebuf, denom, agg);
    // 8: residual+bias, BN statistics
    k_ystats<<<dim3(NN / ROWS_PER_BLK), blk, 0, stream>>>(prev, agg, bias, y, stats);
    // 9: BN normalize + ReLU
    {
        size_t n = (size_t)NN * FF;
        k_bnfinal<<<dim3((unsigned)((n + TPB - 1) / TPB)), blk, 0, stream>>>(y, stats, gamma, beta);
    }
}
